// GroupedMMWrapper_31705448579819
// MI455X (gfx1250) — matrix operand order repaired and checked by reading, not compiled, not run on hardware
//
#include <hip/hip_runtime.h>

// Problem constants (from reference): E=8, K=2048, N=1024, T=4096
#define E_ 8
#define K_ 2048
#define N_ 1024
#define T_ 4096

#define BM 128   // token-tile rows per workgroup
#define BN 128   // output columns per workgroup
#define BK 32    // K step = one bf16 WMMA K
#define LDT 40   // padded LDS row stride in bf16 elems (80B -> conflict-free)

typedef __attribute__((ext_vector_type(8)))  __bf16 v8bf;
typedef __attribute__((ext_vector_type(16))) __bf16 v16bf;
typedef __attribute__((ext_vector_type(4)))  __bf16 v4bf;
typedef __attribute__((ext_vector_type(8)))  float  v8f;

// Split f32 into hi/lo bf16 using native converts (default round mode = RNE).
// x ~= hi + lo to ~16 mantissa bits; 3 bf16 WMMA passes recover ~f32 GEMM.
static __device__ __forceinline__ void split2(float x, __bf16 &h, __bf16 &l) {
  h = (__bf16)x;
  l = (__bf16)(x - (float)h);
}

static __device__ __forceinline__ void split_store(float4 v, __bf16* hp, __bf16* lp) {
  __bf16 h0, l0, h1, l1, h2, l2, h3, l3;
  split2(v.x, h0, l0); split2(v.y, h1, l1);
  split2(v.z, h2, l2); split2(v.w, h3, l3);
  v4bf hv = {h0, h1, h2, h3};
  v4bf lv = {l0, l1, l2, l3};
  *(v4bf*)hp = hv;
  *(v4bf*)lp = lv;
}

// A fragment, 16x32 bf16 (MxK). Per ISA layout: half-wave 0 holds K0-7 (V0-3)
// and K16-23 (V4-7); half-wave 1 holds K8-15 and K24-31. Row M = lane&15.
static __device__ __forceinline__ v16bf frag_a(const __bf16* base, int lane) {
  const int r  = lane & 15;
  const int kh = (lane >> 4) << 3;               // 0 or 8 elems
  const __bf16* p = base + r * LDT + kh;
  v8bf a = *(const v8bf*)(p);
  v8bf b = *(const v8bf*)(p + 16);
  v16bf o;
#pragma unroll
  for (int i = 0; i < 8; ++i) { o[i] = a[i]; o[i + 8] = b[i]; }
  return o;
}

// B fragment, 32x16 bf16 (KxN). Half-wave 0 holds K0-7 (V0-3) and K16-23 (V4-7),
// half-wave 1 holds K8-15 and K24-31. Column N = lane&15. LDS B tile is N-major so this
// is two 16B reads per lane.
static __device__ __forceinline__ v16bf frag_b(const __bf16* base, int lane) {
  const int n  = lane & 15;
  const int kh = (lane >> 4) << 3;               // 0 or 8 elems
  const __bf16* p = base + n * LDT + kh;
  v8bf a = *(const v8bf*)(p);
  v8bf b = *(const v8bf*)(p + 16);
  v16bf o;
#pragma unroll
  for (int i = 0; i < 8; ++i) { o[i] = a[i]; o[i + 8] = b[i]; }
  return o;
}

__global__ __launch_bounds__(256) void grouped_mm_wmma_kernel(
    const float* __restrict__ x,      // [T, K]
    const float* __restrict__ w,      // [E, N, K]
    const int*   __restrict__ offs,   // [E] cumulative ends
    float*       __restrict__ y) {    // [T, N]
  __shared__ __align__(16) __bf16 Ah[BM * LDT];
  __shared__ __align__(16) __bf16 Al[BM * LDT];
  __shared__ __align__(16) __bf16 Bh[BN * LDT];
  __shared__ __align__(16) __bf16 Bl[BN * LDT];

  const int e      = blockIdx.z;
  const int gStart = (e == 0) ? 0 : offs[e - 1];
  const int gEnd   = offs[e];
  const int row0   = gStart + blockIdx.y * BM;
  if (row0 >= gEnd) return;                      // uniform: EXEC stays all-1s
  const int col0   = blockIdx.x * BN;

  const float* we = w + (size_t)e * (size_t)N_ * (size_t)K_;

  const int t    = threadIdx.x;
  const int lane = t & 31;
  const int wv   = t >> 5;
  const int wy   = wv >> 1;                      // 0..3 : 32-row slab
  const int wx   = wv & 1;                       // 0..1 : 64-col slab

  // Loop-invariant cooperative-load descriptors (hoisted addressing+guards).
  const float* xp[4];
  const float* wp[4];
  bool okA[4];
  int  rr[4], rk4[4];
#pragma unroll
  for (int i = 0; i < 4; ++i) {
    int c  = t + i * 256;                        // 0..1023
    rr[i]  = c >> 3;                             // 0..127
    rk4[i] = (c & 7) << 2;                       // 0..28 step 4
    okA[i] = (row0 + rr[i]) < gEnd;
    xp[i]  = x  + (size_t)(row0 + rr[i]) * K_ + rk4[i];
    wp[i]  = we + (size_t)(col0 + rr[i]) * K_ + rk4[i];
  }

  v8f acc[2][4];
#pragma unroll
  for (int mi = 0; mi < 2; ++mi)
#pragma unroll
    for (int ni = 0; ni < 4; ++ni)
#pragma unroll
      for (int i = 0; i < 8; ++i) acc[mi][ni][i] = 0.0f;

  // ---- software pipeline: prefetch k-step 0 into registers ----
  float4 ra[4], rb[4];
#pragma unroll
  for (int i = 0; i < 4; ++i) {
    ra[i] = okA[i] ? *(const float4*)(xp[i]) : make_float4(0.f, 0.f, 0.f, 0.f);
    rb[i] = *(const float4*)(wp[i]);
  }

  for (int kk = 0; kk < K_; kk += BK) {
    __syncthreads();                             // previous tile's readers done

    // ---- convert prefetched regs -> hi/lo bf16 LDS tiles ----
#pragma unroll
    for (int i = 0; i < 4; ++i) {
      const int o = rr[i] * LDT + rk4[i];
      split_store(ra[i], &Ah[o], &Al[o]);
      split_store(rb[i], &Bh[o], &Bl[o]);
    }

    __syncthreads();                             // tile visible

    // ---- prefetch next k-step; latency hidden under the WMMA block ----
    if (kk + BK < K_) {
#pragma unroll
      for (int i = 0; i < 4; ++i) {
        ra[i] = okA[i] ? *(const float4*)(xp[i] + (kk + BK))
                       : make_float4(0.f, 0.f, 0.f, 0.f);
        rb[i] = *(const float4*)(wp[i] + (kk + BK));
      }
    }

    // ---- fragments + WMMA (3x bf16 ~= f32 GEMM) ----
    v16bf ah[2], al[2], bh[4], bl[4];
#pragma unroll
    for (int mi = 0; mi < 2; ++mi) {
      ah[mi] = frag_a(Ah + (wy * 32 + mi * 16) * LDT, lane);
      al[mi] = frag_a(Al + (wy * 32 + mi * 16) * LDT, lane);
    }
#pragma unroll
    for (int ni = 0; ni < 4; ++ni) {
      bh[ni] = frag_b(Bh + (wx * 64 + ni * 16) * LDT, lane);
      bl[ni] = frag_b(Bl + (wx * 64 + ni * 16) * LDT, lane);
    }
#pragma unroll
    for (int mi = 0; mi < 2; ++mi)
#pragma unroll
      for (int ni = 0; ni < 4; ++ni) {
        acc[mi][ni] = __builtin_amdgcn_wmma_f32_16x16x32_bf16(
            false, ah[mi], false, bh[ni], (short)0, acc[mi][ni], false, false);
        acc[mi][ni] = __builtin_amdgcn_wmma_f32_16x16x32_bf16(
            false, al[mi], false, bh[ni], (short)0, acc[mi][ni], false, false);
        acc[mi][ni] = __builtin_amdgcn_wmma_f32_16x16x32_bf16(
            false, ah[mi], false, bl[ni], (short)0, acc[mi][ni], false, false);
      }
  }

  // ---- store C: VGPR v -> M = v + 8*(lane>=16), N = lane&15 ----
  const int ncl = lane & 15;
  const int hs  = lane >> 4;
#pragma unroll
  for (int mi = 0; mi < 2; ++mi)
#pragma unroll
    for (int ni = 0; ni < 4; ++ni) {
      const int col = col0 + wx * 64 + ni * 16 + ncl;
#pragma unroll
      for (int v = 0; v < 8; ++v) {
        const int grow = row0 + wy * 32 + mi * 16 + v + 8 * hs;
        if (grow < gEnd)
          y[(size_t)grow * N_ + col] = acc[mi][ni][v];
      }
    }
}

extern "C" void kernel_launch(void* const* d_in, const int* in_sizes, int n_in,
                              void* d_out, int out_size, void* d_ws, size_t ws_size,
                              hipStream_t stream) {
  (void)in_sizes; (void)n_in; (void)out_size; (void)d_ws; (void)ws_size;
  const float* x    = (const float*)d_in[0];
  const float* w    = (const float*)d_in[1];
  const int*   offs = (const int*)d_in[2];
  float*       y    = (float*)d_out;

  // grid.y covers the worst case (one group owning all T tokens); blocks past
  // a group's end exit uniformly and immediately.
  dim3 grid(N_ / BN, T_ / BM, E_);
  grouped_mm_wmma_kernel<<<grid, 256, 0, stream>>>(x, w, offs, y);
}
